// SABlock_1039382086041
// MI455X (gfx1250) — matrix operand order repaired and checked by reading, not compiled, not run on hardware
//
#include <hip/hip_runtime.h>
#include <hip/hip_bf16.h>

// ---------------------------------------------------------------------------
// Causal multi-head self-attention block for MI455X (gfx1250, wave32, WMMA).
// cvt->bf16 -> QKV WMMA GEMM (unroll-2 reg-pipelined) -> flash attention with
// TDM (tensor_load_to_lds double-buffered K/V tiles, WMMA QK^T + online
// softmax + WMMA PV) -> out-proj WMMA GEMM (+bias, f32).
// ---------------------------------------------------------------------------

typedef __bf16 bf16_t;
typedef __attribute__((ext_vector_type(16))) __bf16 bf16x16;
typedef __attribute__((ext_vector_type(8)))  __bf16 bf16x8;
typedef __attribute__((ext_vector_type(8)))  float  f32x8;
typedef __attribute__((ext_vector_type(4)))  unsigned int u32x4;
typedef __attribute__((ext_vector_type(8)))  unsigned int u32x8;

#define B_  2
#define S_  2048
#define C_  768
#define H_  12
#define D_  64
#define M_  (B_ * S_)          // 4096 flattened rows
#define N3_ (3 * C_)           // 2304 qkv cols
#define SCALE_ 0.125f          // D^-0.5

// ---- WMMA fragment loaders (contiguous k = 16*(lane/16) + i; A and B share this order) ----
static __device__ __forceinline__ bf16x16 load_a_frag(const bf16_t* base, int ld, int lane) {
  const int m  = lane & 15;
  const int hv = lane >> 4;
  const bf16_t* p = base + m * ld + hv * 16;
  bf16x8 lo = *(const bf16x8*)(p);
  bf16x8 hi = *(const bf16x8*)(p + 8);
  bf16x16 a;
#pragma unroll
  for (int i = 0; i < 8; ++i) { a[i] = lo[i]; a[i + 8] = hi[i]; }
  return a;
}

// B 32x16 (KxN) from a TRANSPOSED (row = n, contiguous k) source.
static __device__ __forceinline__ bf16x16 load_b_frag(const bf16_t* bt, int ld, int lane) {
  const int n  = lane & 15;
  const int hv = lane >> 4;
  const bf16_t* p = bt + n * ld + hv * 16;
  bf16x8 lo = *(const bf16x8*)(p);
  bf16x8 hi = *(const bf16x8*)(p + 8);
  bf16x16 b;
#pragma unroll
  for (int i = 0; i < 8; ++i) { b[i] = lo[i]; b[i + 8] = hi[i]; }
  return b;
}

// B 32x16 (KxN) from a ROW-MAJOR [k][n] source (strided gather, used for V).
static __device__ __forceinline__ bf16x16 load_b_frag_strided(const bf16_t* b, int ld, int lane) {
  const int n  = lane & 15;
  const int hv = lane >> 4;
  const bf16_t* p = b + (hv * 16) * ld + n;
  bf16x16 r;
#pragma unroll
  for (int i = 0; i < 16; ++i) r[i] = p[i * ld];
  return r;
}

static __device__ __forceinline__ f32x8 wmma_bf16(bf16x16 a, bf16x16 b, f32x8 c) {
  return __builtin_amdgcn_wmma_f32_16x16x32_bf16(false, a, false, b, (short)0, c, false, false);
}

// ---------------------------------------------------------------------------
// TDM: async-load one 64x64 bf16 tile (row stride 64 elems in memory) into
// LDS with padded stride 80 elems (pad 32B after every 128B row).
// D# layout per CDNA5 ISA. Groups 2/3 zero (dims>2 unused).
// ---------------------------------------------------------------------------
static __device__ __forceinline__ void tdm_load_tile64(unsigned lds_addr, const bf16_t* g) {
  unsigned long long ga = (unsigned long long)(const void*)g;
  u32x4 g0;
  g0[0] = 1u;                                   // count=1, user mode, no gather
  g0[1] = lds_addr;                             // D#.lds_addr
  g0[2] = (unsigned)ga;                         // global_addr[31:0]
  g0[3] = (unsigned)((ga >> 32) & 0x01ffffffull) | (2u << 30);  // addr[56:32], type=2
  u32x8 g1;
  g1[0] = (1u << 16)    // data_size = 1 -> 2 bytes
        | (1u << 20)    // pad_enable
        | (4u << 22)    // pad_interval: 32 DWORDs (=128B) between pads
        | (7u << 25);   // pad_amount: 8 DWORDs (=32B) pad -> LDS stride 160B
  g1[1] = (64u << 16);          // tensor_dim0[15:0] = 64
  g1[2] = (2048u << 16);        // tensor_dim0 hi=0 | tensor_dim1[15:0] = 2048
  g1[3] = (64u << 16);          // tensor_dim1 hi=0 | tile_dim0 = 64
  g1[4] = 64u;                  // tile_dim1 = 64, tile_dim2 = 0
  g1[5] = 64u;                  // tensor_dim0_stride[31:0] = 64 elems
  g1[6] = 0u;                   // stride hi / tensor_dim1_stride lo
  g1[7] = 0u;
  u32x4 gz = (u32x4)(0u);
  asm volatile("tensor_load_to_lds %0, %1, %2, %3"
               :: "s"(g0), "s"(g1), "s"(gz), "s"(gz) : "memory");
}

// ---------------------------------------------------------------------------
// Conversion kernels
// ---------------------------------------------------------------------------
__global__ __launch_bounds__(256) void k_cvt_bf16(const float* __restrict__ src,
                                                  bf16_t* __restrict__ dst, int n) {
  int i = blockIdx.x * 256 + threadIdx.x;
  if (i < n) dst[i] = (bf16_t)src[i];
}

// dst[n*K + k] = src[k*N + n]   (W[k][n] -> Wt[n][k])
__global__ __launch_bounds__(256) void k_cvt_transpose(const float* __restrict__ src,
                                                       bf16_t* __restrict__ dst,
                                                       int K, int N) {
  int i = blockIdx.x * 256 + threadIdx.x;
  if (i < K * N) {
    int nrow = i / K, k = i - nrow * K;
    dst[i] = (bf16_t)src[(size_t)k * N + nrow];
  }
}

// ---------------------------------------------------------------------------
// QKV projection: [4096 x 768] @ [768 x 2304] -> scatter to Q/K/V [b][h][s][d]
// k-loop unrolled by 2 with ping-ponged fragment registers: no copy chain,
// loads for the next 32-wide k-step stay in flight across the 4 WMMAs.
// ---------------------------------------------------------------------------
__global__ __launch_bounds__(128) void k_qkv_gemm(const bf16_t* __restrict__ Xb,
                                                  const bf16_t* __restrict__ Wt,
                                                  bf16_t* __restrict__ Q,
                                                  bf16_t* __restrict__ K,
                                                  bf16_t* __restrict__ V) {
  const int nblk = blockIdx.x, mblk = blockIdx.y;
  const int tid = threadIdx.x, lane = tid & 31, w = tid >> 5;
  const int mbase = mblk * 64 + w * 16;
  const int nbase = nblk * 64;

  f32x8 acc[4];
#pragma unroll
  for (int nb = 0; nb < 4; ++nb) acc[nb] = (f32x8)(0.0f);

  const bf16_t* arow = Xb + (size_t)mbase * C_;
  const bf16_t* brow[4];
#pragma unroll
  for (int nb = 0; nb < 4; ++nb) brow[nb] = Wt + (size_t)(nbase + nb * 16) * C_;

  bf16x16 a0 = load_a_frag(arow, C_, lane);
  bf16x16 b0[4];
#pragma unroll
  for (int nb = 0; nb < 4; ++nb) b0[nb] = load_b_frag(brow[nb], C_, lane);

  for (int k = 0; k < C_; k += 64) {          // 12 double-steps (C_ % 64 == 0)
    // stage A: prefetch k+32 (always in range), compute on set 0
    bf16x16 a1 = load_a_frag(arow + k + 32, C_, lane);
    bf16x16 b1[4];
#pragma unroll
    for (int nb = 0; nb < 4; ++nb) b1[nb] = load_b_frag(brow[nb] + k + 32, C_, lane);
#pragma unroll
    for (int nb = 0; nb < 4; ++nb) acc[nb] = wmma_bf16(a0, b0[nb], acc[nb]);
    // stage B: prefetch k+64 (skip on last), compute on set 1
    if (k + 64 < C_) {
      a0 = load_a_frag(arow + k + 64, C_, lane);
#pragma unroll
      for (int nb = 0; nb < 4; ++nb) b0[nb] = load_b_frag(brow[nb] + k + 64, C_, lane);
    }
#pragma unroll
    for (int nb = 0; nb < 4; ++nb) acc[nb] = wmma_bf16(a1, b1[nb], acc[nb]);
  }

  const int n0 = lane & 15, hv = lane >> 4;
#pragma unroll
  for (int nb = 0; nb < 4; ++nb) {
    int ng  = nbase + nb * 16 + n0;
    int sel = ng / C_;
    int rem = ng - sel * C_;
    int h   = rem >> 6;
    int d   = rem & 63;
    bf16_t* dst = (sel == 0) ? Q : (sel == 1) ? K : V;
#pragma unroll
    for (int r = 0; r < 8; ++r) {
      int mg = mbase + r + 8 * hv;
      int b  = mg >> 11;
      int s  = mg & (S_ - 1);
      dst[((size_t)((b * H_ + h) * S_ + s)) * D_ + d] = (bf16_t)acc[nb][r];
    }
  }
}

// ---------------------------------------------------------------------------
// Flash attention: grid = (S/64 q-tiles, B*H). Block = 4 waves, each owns 16
// queries. K/V tiles (64 keys) DMA'd into double-buffered LDS by the TDM,
// prefetch of tile kt+1 overlaps WMMA+softmax of tile kt.
// ---------------------------------------------------------------------------
#define LDSD 80   // padded LDS leading dim (bf16 elems); 160B keeps 16B align

__global__ __launch_bounds__(128) void k_attn(const bf16_t* __restrict__ Q,
                                              const bf16_t* __restrict__ K,
                                              const bf16_t* __restrict__ V,
                                              bf16_t* __restrict__ att) {
  const int qt  = blockIdx.x;           // query tile (0..31)
  const int bh  = blockIdx.y;           // b*H + h   (0..23)
  const int b   = bh / H_, h = bh - b * H_;
  const int tid = threadIdx.x, lane = tid & 31, w = tid >> 5;
  const int n0  = lane & 15, hv = lane >> 4;

  __shared__ bf16_t sK[2][64 * LDSD];   // K tile  [key][d]   (double-buffered)
  __shared__ bf16_t sV[2][64 * LDSD];   // V tile  [key][d]   (double-buffered)
  __shared__ bf16_t sP[4 * 16 * LDSD];  // per-wave probability tile [m][key]

  const size_t head_off = (size_t)bh * S_ * D_;
  const bf16_t* Qh = Q + head_off;
  const bf16_t* Kh = K + head_off;
  const bf16_t* Vh = V + head_off;

  const unsigned ldsK[2] = { (unsigned)(size_t)&sK[0][0], (unsigned)(size_t)&sK[1][0] };
  const unsigned ldsV[2] = { (unsigned)(size_t)&sV[0][0], (unsigned)(size_t)&sV[1][0] };

  const int qbase = qt * 64 + w * 16;
  bf16x16 aq0 = load_a_frag(Qh + (size_t)qbase * D_ + 0,  D_, lane);
  bf16x16 aq1 = load_a_frag(Qh + (size_t)qbase * D_ + 32, D_, lane);

  f32x8 O[4];
#pragma unroll
  for (int nb = 0; nb < 4; ++nb) O[nb] = (f32x8)(0.0f);
  float rmax[8], rsum[8];
#pragma unroll
  for (int r = 0; r < 8; ++r) { rmax[r] = -1e30f; rsum[r] = 0.0f; }

  bf16_t* Pw = &sP[w * 16 * LDSD];
  const int nkt = qt + 1;               // causal: key tiles 0..qt

  // prologue: DMA tile 0 into buffer 0 (one TDM pair, issued by wave 0)
  if (w == 0) {
    tdm_load_tile64(ldsK[0], Kh);
    tdm_load_tile64(ldsV[0], Vh);
  }

  for (int kt = 0; kt < nkt; ++kt) {
    const int buf = kt & 1;
    // tile kt's DMA must have landed (wave0 only has tile kt outstanding).
    __builtin_amdgcn_s_wait_tensorcnt(0);
    __syncthreads();                    // tile kt visible to all waves

    // prefetch tile kt+1 into the other buffer (overlaps compute below);
    // safe: all waves finished reading buf^1 before the barrier above.
    if (w == 0 && kt + 1 < nkt) {
      const size_t off = (size_t)(kt + 1) * 64 * D_;
      tdm_load_tile64(ldsK[buf ^ 1], Kh + off);
      tdm_load_tile64(ldsV[buf ^ 1], Vh + off);
    }

    const int kbase = kt * 64;
    const bf16_t* sKb = &sK[buf][0];
    const bf16_t* sVb = &sV[buf][0];

    // ---- S = Q @ K^T  (4 col-blocks x 2 k-steps) ----
    f32x8 Sf[4];
#pragma unroll
    for (int nb = 0; nb < 4; ++nb) {
      f32x8 c = (f32x8)(0.0f);
      c = wmma_bf16(aq0, load_b_frag(sKb + (nb * 16) * LDSD + 0,  LDSD, lane), c);
      c = wmma_bf16(aq1, load_b_frag(sKb + (nb * 16) * LDSD + 32, LDSD, lane), c);
      Sf[nb] = c;
    }

    // ---- mask + scale + tile row max ----
    float tmax[8];
#pragma unroll
    for (int r = 0; r < 8; ++r) tmax[r] = -1e30f;
#pragma unroll
    for (int nb = 0; nb < 4; ++nb) {
      int kcol = kbase + nb * 16 + n0;
#pragma unroll
      for (int r = 0; r < 8; ++r) {
        int qrow = qbase + r + 8 * hv;
        float s = (kcol <= qrow) ? Sf[nb][r] * SCALE_ : -1e30f;
        Sf[nb][r] = s;
        tmax[r] = fmaxf(tmax[r], s);
      }
    }
#pragma unroll
    for (int r = 0; r < 8; ++r) {
#pragma unroll
      for (int msk = 8; msk >= 1; msk >>= 1)
        tmax[r] = fmaxf(tmax[r], __shfl_xor(tmax[r], msk, 16));
    }

    // ---- online softmax update ----
    float corr[8];
#pragma unroll
    for (int r = 0; r < 8; ++r) {
      float nm = fmaxf(rmax[r], tmax[r]);
      corr[r] = __expf(rmax[r] - nm);
      rmax[r] = nm;
    }
    float tsum[8];
#pragma unroll
    for (int r = 0; r < 8; ++r) tsum[r] = 0.0f;
#pragma unroll
    for (int nb = 0; nb < 4; ++nb) {
#pragma unroll
      for (int r = 0; r < 8; ++r) {
        float p = __expf(Sf[nb][r] - rmax[r]);
        tsum[r] += p;
        Pw[(r + 8 * hv) * LDSD + nb * 16 + n0] = (bf16_t)p;   // wave-local LDS
      }
    }
#pragma unroll
    for (int r = 0; r < 8; ++r) rsum[r] = rsum[r] * corr[r] + tsum[r];
#pragma unroll
    for (int nb = 0; nb < 4; ++nb)
#pragma unroll
      for (int r = 0; r < 8; ++r) O[nb][r] *= corr[r];

    // ---- O += P @ V  (same-wave LDS ops are in-order; V gathered strided) ----
#pragma unroll
    for (int ks = 0; ks < 2; ++ks) {
      bf16x16 ap = load_a_frag(Pw + ks * 32, LDSD, lane);
#pragma unroll
      for (int nb = 0; nb < 4; ++nb) {
        bf16x16 bv = load_b_frag_strided(sVb + (ks * 32) * LDSD + nb * 16, LDSD, lane);
        O[nb] = wmma_bf16(ap, bv, O[nb]);
      }
    }
  }

  // ---- normalize, store att[b*S+q][h*64+d] (bf16, ld = C) ----
#pragma unroll
  for (int nb = 0; nb < 4; ++nb) {
#pragma unroll
    for (int r = 0; r < 8; ++r) {
      int q = qbase + r + 8 * hv;
      att[((size_t)(b * S_ + q)) * C_ + h * D_ + nb * 16 + n0] =
          (bf16_t)(O[nb][r] / rsum[r]);
    }
  }
}

// ---------------------------------------------------------------------------
// Output projection: [4096 x 768] @ [768 x 768] + bias -> f32 d_out
// (same unroll-2 ping-pong pipeline as the QKV GEMM)
// ---------------------------------------------------------------------------
__global__ __launch_bounds__(128) void k_out_gemm(const bf16_t* __restrict__ A,
                                                  const bf16_t* __restrict__ Wt,
                                                  const float* __restrict__ bias,
                                                  float* __restrict__ out) {
  const int nblk = blockIdx.x, mblk = blockIdx.y;
  const int tid = threadIdx.x, lane = tid & 31, w = tid >> 5;
  const int mbase = mblk * 64 + w * 16;
  const int nbase = nblk * 64;

  f32x8 acc[4];
#pragma unroll
  for (int nb = 0; nb < 4; ++nb) acc[nb] = (f32x8)(0.0f);

  const bf16_t* arow = A + (size_t)mbase * C_;
  const bf16_t* brow[4];
#pragma unroll
  for (int nb = 0; nb < 4; ++nb) brow[nb] = Wt + (size_t)(nbase + nb * 16) * C_;

  bf16x16 a0 = load_a_frag(arow, C_, lane);
  bf16x16 b0[4];
#pragma unroll
  for (int nb = 0; nb < 4; ++nb) b0[nb] = load_b_frag(brow[nb], C_, lane);

  for (int k = 0; k < C_; k += 64) {
    bf16x16 a1 = load_a_frag(arow + k + 32, C_, lane);
    bf16x16 b1[4];
#pragma unroll
    for (int nb = 0; nb < 4; ++nb) b1[nb] = load_b_frag(brow[nb] + k + 32, C_, lane);
#pragma unroll
    for (int nb = 0; nb < 4; ++nb) acc[nb] = wmma_bf16(a0, b0[nb], acc[nb]);
    if (k + 64 < C_) {
      a0 = load_a_frag(arow + k + 64, C_, lane);
#pragma unroll
      for (int nb = 0; nb < 4; ++nb) b0[nb] = load_b_frag(brow[nb] + k + 64, C_, lane);
    }
#pragma unroll
    for (int nb = 0; nb < 4; ++nb) acc[nb] = wmma_bf16(a1, b1[nb], acc[nb]);
  }

  const int n0 = lane & 15, hv = lane >> 4;
#pragma unroll
  for (int nb = 0; nb < 4; ++nb) {
    int ng = nbase + nb * 16 + n0;
    float bv = bias[ng];
#pragma unroll
    for (int r = 0; r < 8; ++r) {
      int mg = mbase + r + 8 * hv;
      out[(size_t)mg * C_ + ng] = acc[nb][r] + bv;
    }
  }
}

// ---------------------------------------------------------------------------
extern "C" void kernel_launch(void* const* d_in, const int* in_sizes, int n_in,
                              void* d_out, int out_size, void* d_ws, size_t ws_size,
                              hipStream_t stream) {
  const float* x     = (const float*)d_in[0];   // [B,S,C]
  const float* Wqkv  = (const float*)d_in[1];   // [C,3C]
  const float* Wout  = (const float*)d_in[2];   // [C,C]
  const float* bout  = (const float*)d_in[3];   // [C]

  char* ws = (char*)d_ws;
  size_t off = 0;
  bf16_t* Xb    = (bf16_t*)(ws + off); off += (size_t)M_  * C_  * 2;
  bf16_t* Wqt   = (bf16_t*)(ws + off); off += (size_t)N3_ * C_  * 2;
  bf16_t* Wot   = (bf16_t*)(ws + off); off += (size_t)C_  * C_  * 2;
  bf16_t* Qb    = (bf16_t*)(ws + off); off += (size_t)M_  * C_  * 2;
  bf16_t* Kb    = (bf16_t*)(ws + off); off += (size_t)M_  * C_  * 2;
  bf16_t* Vb    = (bf16_t*)(ws + off); off += (size_t)M_  * C_  * 2;
  bf16_t* attb  = (bf16_t*)(ws + off); off += (size_t)M_  * C_  * 2;

  // 1) precision conversion / weight transpose (bf16, Wt[n][k])
  k_cvt_bf16     <<<(M_ * C_ + 255) / 256, 256, 0, stream>>>(x, Xb, M_ * C_);
  k_cvt_transpose<<<(C_ * N3_ + 255) / 256, 256, 0, stream>>>(Wqkv, Wqt, C_, N3_);
  k_cvt_transpose<<<(C_ * C_  + 255) / 256, 256, 0, stream>>>(Wout, Wot, C_, C_);

  // 2) QKV projection -> Q/K/V [b][h][s][d]
  k_qkv_gemm<<<dim3(N3_ / 64, M_ / 64), 128, 0, stream>>>(Xb, Wqt, Qb, Kb, Vb);

  // 3) causal flash attention (TDM-staged tiles) -> att [b*S+s][h*D+d]
  k_attn<<<dim3(S_ / 64, B_ * H_), 128, 0, stream>>>(Qb, Kb, Vb, attb);

  // 4) output projection + bias -> f32 d_out
  k_out_gemm<<<dim3(C_ / 64, M_ / 64), 128, 0, stream>>>(attb, Wot, bout, (float*)d_out);
}
